// VectorQuantizer_78941498900941
// MI455X (gfx1250) — matrix operand order repaired and checked by reading, not compiled, not run on hardware
//
#include <hip/hip_runtime.h>

#define CDIM 256
#define KCODES 1024
#define NROWS 32768
#define BM 128
#define HW 1024
#define OUT_ELEMS (NROWS * CDIM) /* 8388608 */

typedef _Float16 h16v __attribute__((ext_vector_type(16)));
typedef float f4v __attribute__((ext_vector_type(4)));
typedef float f8v __attribute__((ext_vector_type(8)));

union Frag { h16v h; f4v f[2]; };

__device__ __forceinline__ unsigned pack2h(_Float16 a, _Float16 b) {
    union { _Float16 h[2]; unsigned u; } p;
    p.h[0] = a; p.h[1] = b; return p.u;
}

// ---------------- prep: split embeddings to f16 hi/lo, ||e||^2, zero loss ----
__global__ __launch_bounds__(256) void vq_prep_split(
    const float* __restrict__ emb, _Float16* __restrict__ Eh,
    _Float16* __restrict__ El, float* __restrict__ enorm2,
    float* __restrict__ loss_slot) {
    __shared__ float red[256];
    const int k = blockIdx.x;
    const int c = threadIdx.x;
    float e = emb[k * CDIM + c];
    _Float16 h = (_Float16)e;
    _Float16 l = (_Float16)(e - (float)h);
    Eh[k * CDIM + c] = h;
    El[k * CDIM + c] = l;
    red[c] = e * e;
    __syncthreads();
    for (int s = 128; s > 0; s >>= 1) {
        if (c < s) red[c] += red[c + s];
        __syncthreads();
    }
    if (c == 0) enorm2[k] = red[0];
    if (k == 0 && c == 0) *loss_slot = 0.0f;
}

// ---------------- main: split-f16 WMMA distance GEMM + fused argmin + gather --
__global__ __launch_bounds__(256) void vq_main(
    const float* __restrict__ x, const float* __restrict__ emb,
    const _Float16* __restrict__ Eh, const _Float16* __restrict__ El,
    const float* __restrict__ enorm2, float* __restrict__ out,
    float* __restrict__ loss_slot) {

    // A slab: 128 rows x 256 ch, row stride 264 halfs (16B pad -> conflict-free b128 reads)
    __shared__ _Float16 AhS[BM * 264];
    __shared__ _Float16 AlS[BM * 264];
    // B tile: 128 codes x 32 ch, row stride 40 halfs (16B pad)
    __shared__ _Float16 BhS[128 * 40];
    __shared__ _Float16 BlS[128 * 40];
    __shared__ float eLDS[KCODES];

    const int tid  = threadIdx.x;
    const int wave = tid >> 5;
    const int lane = tid & 31;
    const int l16  = lane & 15;
    const int lhi  = lane >> 4; // 0 or 1

    const int n0  = blockIdx.x * BM;
    const int bb  = n0 / HW;   // batch index (BM divides HW)
    const int hw0 = n0 % HW;

    // ---- stage A (x rows) as split f16 ----
    {
        unsigned* AhU = (unsigned*)AhS;
        unsigned* AlU = (unsigned*)AlS;
        const int nl  = tid & 127;
        const int c2b = tid >> 7;          // 0..1
        for (int c2 = c2b; c2 < CDIM / 2; c2 += 2) {
            const int c = c2 * 2;
            float x0 = x[(bb * CDIM + c) * HW + hw0 + nl];
            float x1 = x[(bb * CDIM + c + 1) * HW + hw0 + nl];
            _Float16 h0 = (_Float16)x0; _Float16 l0 = (_Float16)(x0 - (float)h0);
            _Float16 h1 = (_Float16)x1; _Float16 l1 = (_Float16)(x1 - (float)h1);
            AhU[nl * 132 + c2] = pack2h(h0, h1);
            AlU[nl * 132 + c2] = pack2h(l0, l1);
        }
        for (int i = tid; i < KCODES; i += 256) eLDS[i] = enorm2[i];
    }

    float bval[8];
    int   bidx[8];
#pragma unroll
    for (int r = 0; r < 8; ++r) { bval[r] = 3.4e38f; bidx[r] = 0; }

    const unsigned* EhU = (const unsigned*)Eh;
    const unsigned* ElU = (const unsigned*)El;
    unsigned* BhU = (unsigned*)BhS;
    unsigned* BlU = (unsigned*)BlS;

    for (int cg = 0; cg < 8; ++cg) {          // 8 column groups of 128 codes
        f8v acc[8];
#pragma unroll
        for (int t = 0; t < 8; ++t) {
#pragma unroll
            for (int e = 0; e < 8; ++e) acc[t][e] = 0.0f;
        }

        for (int kt = 0; kt < 8; ++kt) {      // K loop: 8 x 32 channels
            __syncthreads();
            // stage B tile (hi+lo) from pre-split codebook
            for (int p = tid; p < 2048; p += 256) {
                const int j  = p >> 4;
                const int c2 = p & 15;
                const int gi = (cg * 128 + j) * (CDIM / 2) + kt * 16 + c2;
                BhU[j * 20 + c2] = EhU[gi];
                BlU[j * 20 + c2] = ElU[gi];
            }
            __syncthreads();

            // A fragments for this wave's 16-row m-tile (documented 16-bit A layout)
            const int rowm  = wave * 16 + l16;
            const int abase = rowm * 264 + kt * 32 + lhi * 8;
            Frag ah, al;
            ah.f[0] = *(const f4v*)&AhS[abase];
            ah.f[1] = *(const f4v*)&AhS[abase + 16];
            al.f[0] = *(const f4v*)&AlS[abase];
            al.f[1] = *(const f4v*)&AlS[abase + 16];

#pragma unroll
            for (int t = 0; t < 8; ++t) {
                const int bbase = (t * 16 + l16) * 40 + lhi * 8;
                Frag bh, bl;
                bh.f[0] = *(const f4v*)&BhS[bbase];
                bh.f[1] = *(const f4v*)&BhS[bbase + 16];
                bl.f[0] = *(const f4v*)&BlS[bbase];
                bl.f[1] = *(const f4v*)&BlS[bbase + 16];
                // split-f32 product: xh*eh + xh*el + xl*eh (error ~2^-22)
                acc[t] = __builtin_amdgcn_wmma_f32_16x16x32_f16(
                    false, ah.h, false, bh.h, (short)0, acc[t], false, false);
                acc[t] = __builtin_amdgcn_wmma_f32_16x16x32_f16(
                    false, ah.h, false, bl.h, (short)0, acc[t], false, false);
                acc[t] = __builtin_amdgcn_wmma_f32_16x16x32_f16(
                    false, al.h, false, bh.h, (short)0, acc[t], false, false);
            }
        }

        // fused argmin update: score = ||e_k||^2 - 2*S (||x||^2 constant per row)
#pragma unroll
        for (int t = 0; t < 8; ++t) {
            const int col = cg * 128 + t * 16 + l16;
            const float en = eLDS[col];
#pragma unroll
            for (int r = 0; r < 8; ++r) {
                float s = en - 2.0f * acc[t][r];
                if (s < bval[r]) { bval[r] = s; bidx[r] = col; }
            }
        }
    }

    __syncthreads();                 // B tiles dead: reuse their LDS
    int*   rowbest = (int*)BhS;      // 128 ints
    float* red     = (float*)&BhS[256]; // 256 floats

    // cross-lane (16-lane group) argmin; lane group holds one row's column subsets
#pragma unroll
    for (int r = 0; r < 8; ++r) {
        float v = bval[r]; int ix = bidx[r];
#pragma unroll
        for (int m = 1; m < 16; m <<= 1) {
            float ov = __shfl_xor(v, m, 16);
            int   oi = __shfl_xor(ix, m, 16);
            if (ov < v || (ov == v && oi < ix)) { v = ov; ix = oi; }
        }
        if (l16 == 0) rowbest[wave * 16 + lhi * 8 + r] = ix;
    }
    __syncthreads();

    // ---- gather quantized output (coalesced along HW) + loss partial ----
    float lacc = 0.0f;
    for (int i = tid; i < BM * CDIM; i += 256) {
        const int n   = i & (BM - 1);
        const int c   = i >> 7;
        const int kst = rowbest[n];
        const float e  = emb[kst * CDIM + c];
        const int oi   = (bb * CDIM + c) * HW + hw0 + n;
        const float xv = x[oi];
        out[oi] = e;
        const float d = e - xv;
        lacc += d * d;
    }
    red[tid] = lacc;
    __syncthreads();
    for (int s = 128; s > 0; s >>= 1) {
        if (tid < s) red[tid] += red[tid + s];
        __syncthreads();
    }
    // loss = (1 + 0.25) * mean((q - x)^2)
    if (tid == 0) atomicAdd(loss_slot, red[0] * (1.25f / (float)OUT_ELEMS));
}

extern "C" void kernel_launch(void* const* d_in, const int* in_sizes, int n_in,
                              void* d_out, int out_size, void* d_ws, size_t ws_size,
                              hipStream_t stream) {
    const float* x   = (const float*)d_in[0];   // [32,256,32,32]
    const float* emb = (const float*)d_in[1];   // [1024,256]
    float* out  = (float*)d_out;                // 8388608 + 1 (loss)
    float* loss = out + OUT_ELEMS;

    // workspace: Eh (512KB) | El (512KB) | enorm2 (4KB)
    _Float16* Eh = (_Float16*)d_ws;
    _Float16* El = Eh + KCODES * CDIM;
    float* enorm2 = (float*)(El + KCODES * CDIM);

    vq_prep_split<<<KCODES, 256, 0, stream>>>(emb, Eh, El, enorm2, loss);
    vq_main<<<NROWS / BM, 256, 0, stream>>>(x, emb, Eh, El, enorm2, out, loss);
}
